// FastWeightMemory_47708496724074
// MI455X (gfx1250) — matrix operand order repaired and checked by reading, not compiled, not run on hardware
//
#include <hip/hip_runtime.h>

// Problem constants (from reference setup_inputs)
#define BB    4
#define SS    4096
#define HH    1024
#define MD    256
#define CC    64
#define NCHK  64          // SS / CC
#define RTOT  16384       // BB * SS

typedef __attribute__((ext_vector_type(16))) __bf16         v16bf;
typedef __attribute__((ext_vector_type(8)))  float          v8f;
typedef __attribute__((ext_vector_type(4)))  float          float4v;
typedef __attribute__((ext_vector_type(8)))  unsigned short ushort8;
typedef __attribute__((ext_vector_type(4)))  unsigned short ushort4v;
typedef __attribute__((ext_vector_type(4)))  unsigned int   uint4v;
typedef __attribute__((ext_vector_type(8)))  int            int8v;
typedef __attribute__((ext_vector_type(4)))  int            int4v;

#if defined(__has_builtin)
#  if __has_builtin(__builtin_amdgcn_tensor_load_to_lds)
#    define HAVE_TDM 1
#  endif
#endif
#ifndef HAVE_TDM
#  define HAVE_TDM 0
#endif

union BF16Frag {
  v16bf v;
  unsigned short u[16];
  ushort8 h[2];
};

__device__ __forceinline__ unsigned short f2bf_u(float f) {
  unsigned u = __builtin_bit_cast(unsigned, f);
  u += 0x7FFFu + ((u >> 16) & 1u);      // round-to-nearest-even
  return (unsigned short)(u >> 16);
}

// B fragment (32x16 bf16, K=32): lane holds col n = lane&15; element e <-> K = e + 16*hi.
// Source is the transposed operand stored row-major (N x K): 16 contiguous halves.
__device__ __forceinline__ v16bf load_bfrag(const unsigned short* p) {
  BF16Frag r;
  r.h[0] = *(const ushort8*)(p);
  r.h[1] = *(const ushort8*)(p + 8);
  return r.v;
}

// A fragment (16x32 bf16): lane holds row m = lane&15; K = e + 16*hi.
// Row-major bf16 source (global or LDS).
__device__ __forceinline__ v16bf load_afrag_bf(const unsigned short* row, int hi) {
  BF16Frag r;
  r.h[0] = *(const ushort8*)(row + 16 * hi);
  r.h[1] = *(const ushort8*)(row + 16 * hi + 8);
  return r.v;
}

__device__ __forceinline__ v8f wmma_bf16(v16bf a, v16bf b, v8f c) {
  return __builtin_amdgcn_wmma_f32_16x16x32_bf16(false, a, false, b, (short)0, c,
                                                 false, false);
}

// LDS byte offset of a __shared__ object (generic-pointer low 32 bits = LDS offset).
__device__ __forceinline__ unsigned lds_offset(const void* p) {
  return (unsigned)(uintptr_t)p;
}

#if HAVE_TDM
// TDM: 1-D contiguous tile of `elems` bf16 elements from global -> LDS.
// D# per CDNA5 ISA ch.8: group0 = {count=1, lds_addr, global_addr[56:0], type=2};
// group1 = {data_size=2B, tensor_dim0=elems, tensor_dim1=1, tile_dim0=elems,
//           tile_dim1=1, dim0_stride=elems}.  elems must be <= 65535.
// amdgpu-toolchain (clang-23) builtin: 6 args
//   (uint32x4 g0, int32x8 g1, int32x4 g2, int32x4 g3, int32x8 pad, i32 cpol)
__device__ __forceinline__ void tdm_load_1d(const void* gsrc, unsigned ldsOff,
                                            unsigned elems) {
  size_t ga = (size_t)gsrc;
  uint4v g0;
  int8v  g1;
  int4v  z4 = {};
  int8v  z8 = {};
  g0[0] = 1u;                                              // count=1, user mode
  g0[1] = ldsOff;                                          // lds_addr (bytes)
  g0[2] = (unsigned)ga;                                    // global_addr[31:0]
  g0[3] = ((unsigned)(ga >> 32) & 0x01FFFFFFu) | (2u << 30); // addr[56:32], type=2
  g1[0] = 0x00010000;                                      // data_size = 1 (2 bytes)
  g1[1] = (int)((elems & 0xFFFFu) << 16);                  // tensor_dim0[15:0]
  g1[2] = (int)(((elems >> 16) & 0xFFFFu) | (1u << 16));   // dim0[31:16], dim1=1
  g1[3] = (int)((elems & 0xFFFFu) << 16);                  // dim1 hi=0, tile_dim0
  g1[4] = 1;                                               // tile_dim1=1, tile_dim2=0
  g1[5] = (int)elems;                                      // dim0_stride (unused, 1 row)
  g1[6] = 0;
  g1[7] = 0;
  __builtin_amdgcn_tensor_load_to_lds(g0, g1, z4, z4, z8, 0);
}
#endif

// Stage `elems` bf16 (contiguous) into LDS; whole block participates.
__device__ __forceinline__ void stage_tile(unsigned short* sdst,
                                           const unsigned short* gsrc,
                                           unsigned elems, int tid, int wave) {
#if HAVE_TDM
  if (wave == 0) {
    tdm_load_1d(gsrc, lds_offset(sdst), elems);
    __builtin_amdgcn_s_wait_tensorcnt(0);
  }
  (void)tid;
#else
  for (unsigned i = tid; i < elems / 8; i += 256)
    ((ushort8*)sdst)[i] = ((const ushort8*)gsrc)[i];
#endif
  __syncthreads();
}

// ---------------------------------------------------------------------------
// Kernel 0a: f32 -> bf16 conversion, 4 elements/thread (n % 4 == 0)
// ---------------------------------------------------------------------------
__global__ __launch_bounds__(256) void cvt_f32_bf16(const float* __restrict__ src,
                                                    unsigned short* __restrict__ dst,
                                                    int n) {
  int i = (blockIdx.x * 256 + threadIdx.x) * 4;
  if (i < n) {
    float4v v = *(const float4v*)(src + i);
    ushort4v o;
    o[0] = f2bf_u(v[0]); o[1] = f2bf_u(v[1]);
    o[2] = f2bf_u(v[2]); o[3] = f2bf_u(v[3]);
    *(ushort4v*)(dst + i) = o;
  }
}

// ---------------------------------------------------------------------------
// Kernel 1: fused q/k/v projection + L2 norm of k,v.
// Block = 16 permuted tokens x 256 features; A tile (16x1024 bf16, 32KB) staged
// once into LDS via TDM; 8 waves x 2 col tiles x 3 outputs.
// Permutation: p = ck*256 + b*64 + j, with s = ck*64 + j.
// ---------------------------------------------------------------------------
__global__ __launch_bounds__(256) void proj_qkv(
    const unsigned short* __restrict__ xbf,   // RTOT x HH bf16, row-major (b,s)
    const unsigned short* __restrict__ WqBf,
    const unsigned short* __restrict__ WkBf,
    const unsigned short* __restrict__ WvBf,
    unsigned short* __restrict__ qOut,   // RTOT x MD, row-major (permuted)
    unsigned short* __restrict__ kT,     // MD x RTOT (transposed, permuted)
    unsigned short* __restrict__ vT) {   // MD x RTOT
  __shared__ unsigned short sTile[16 * HH];   // 32 KB A tile
  __shared__ float ssum[2][16];               // [0]=k sumsq per row, [1]=v

  const int tid  = threadIdx.x;
  const int wave = tid >> 5;
  const int lane = tid & 31;
  const int hi   = lane >> 4;
  const int ln   = lane & 15;

  const int P     = blockIdx.x * 16;             // permuted token base
  const int b     = (P >> 6) & 3;
  const int sBase = (P >> 8) * 64 + (P & 63);    // contiguous 16 s-values

  if (tid < 32) ssum[tid >> 4][tid & 15] = 0.0f;
  stage_tile(sTile, xbf + (size_t)(b * SS + sBase) * HH, 16 * HH, tid, wave);

  const int n0 = wave * 32 + ln;
  const int n1 = n0 + 16;
  const unsigned short* wq0 = WqBf + (size_t)n0 * HH;
  const unsigned short* wq1 = WqBf + (size_t)n1 * HH;
  const unsigned short* wk0 = WkBf + (size_t)n0 * HH;
  const unsigned short* wk1 = WkBf + (size_t)n1 * HH;
  const unsigned short* wv0 = WvBf + (size_t)n0 * HH;
  const unsigned short* wv1 = WvBf + (size_t)n1 * HH;

  v8f accQ0{}, accQ1{}, accK0{}, accK1{}, accV0{}, accV1{};

  for (int k0 = 0; k0 < HH; k0 += 32) {
    v16bf a = load_afrag_bf(&sTile[ln * HH + k0], hi);   // ds_load_b128 x2
    const int off = k0 + 16 * hi;
    accQ0 = wmma_bf16(a, load_bfrag(wq0 + off), accQ0);
    accQ1 = wmma_bf16(a, load_bfrag(wq1 + off), accQ1);
    accK0 = wmma_bf16(a, load_bfrag(wk0 + off), accK0);
    accK1 = wmma_bf16(a, load_bfrag(wk1 + off), accK1);
    accV0 = wmma_bf16(a, load_bfrag(wv0 + off), accV0);
    accV1 = wmma_bf16(a, load_bfrag(wv1 + off), accV1);
  }

  __syncthreads();
#pragma unroll
  for (int i = 0; i < 8; ++i) {
    const int r = i + 8 * hi;
    atomicAdd(&ssum[0][r], accK0[i] * accK0[i] + accK1[i] * accK1[i]);
    atomicAdd(&ssum[1][r], accV0[i] * accV0[i] + accV1[i] * accV1[i]);
  }
  __syncthreads();

  // store q (row-major, strided scalar stores)
#pragma unroll
  for (int i = 0; i < 8; ++i) {
    const size_t row = (size_t)(P + i + 8 * hi) * MD;
    qOut[row + n0] = f2bf_u(accQ0[i]);
    qOut[row + n1] = f2bf_u(accQ1[i]);
  }

  // store normalized k,v transposed (contiguous 16B per lane per tile)
  ushort8 ok0, ok1, ov0, ov1;
#pragma unroll
  for (int i = 0; i < 8; ++i) {
    const int r = i + 8 * hi;
    const float sk = 1.0f / fmaxf(sqrtf(ssum[0][r]), 1e-12f);
    const float sv = 1.0f / fmaxf(sqrtf(ssum[1][r]), 1e-12f);
    ok0[i] = f2bf_u(accK0[i] * sk);
    ok1[i] = f2bf_u(accK1[i] * sk);
    ov0[i] = f2bf_u(accV0[i] * sv);
    ov1[i] = f2bf_u(accV1[i] * sv);
  }
  *(ushort8*)(kT + (size_t)n0 * RTOT + P + 8 * hi) = ok0;
  *(ushort8*)(kT + (size_t)n1 * RTOT + P + 8 * hi) = ok1;
  *(ushort8*)(vT + (size_t)n0 * RTOT + P + 8 * hi) = ov0;
  *(ushort8*)(vT + (size_t)n1 * RTOT + P + 8 * hi) = ov1;
}

// ---------------------------------------------------------------------------
// Kernel 2: sequential state recurrence. One workgroup; M lives in 256KB LDS.
// Per chunk: snapshot bf16(M); M = 0.99*M + (vT_ck @ k_ck)/256; clamp ||M||_F <= 10.
// ---------------------------------------------------------------------------
__global__ __launch_bounds__(256) void scan_states(
    const float* __restrict__ M0,
    const unsigned short* __restrict__ kT,
    const unsigned short* __restrict__ vT,
    unsigned short* __restrict__ Mstates) {  // NCHK x 65536 bf16
  extern __shared__ float Msh[];             // 65536 state + [65536] norm accum

  const int tid  = threadIdx.x;
  const int wave = tid >> 5;
  const int lane = tid & 31;
  const int hi   = lane >> 4;
  const int ln   = lane & 15;

  for (int i = tid; i < 65536; i += 256) Msh[i] = M0[i];
  __syncthreads();

  for (int ck = 0; ck < NCHK; ++ck) {
    // snapshot state at start of chunk
    unsigned short* dst = Mstates + (size_t)ck * 65536;
    for (int i = tid; i < 65536; i += 256) dst[i] = f2bf_u(Msh[i]);
    __syncthreads();

    // outer = vT_ck @ k_ck  (256x256, K = 256 chunk tokens); 256 tiles / 8 waves
    for (int t = 0; t < 32; ++t) {
      const int tile = wave * 32 + t;
      const int ti = tile >> 4;
      const int tj = tile & 15;
      const unsigned short* arow = vT + (size_t)(ti * 16 + ln) * RTOT + ck * 256;
      const unsigned short* brow = kT + (size_t)(tj * 16 + ln) * RTOT + ck * 256;
      v8f acc{};
      for (int k0 = 0; k0 < 256; k0 += 32)
        acc = wmma_bf16(load_afrag_bf(arow + k0, hi),
                        load_bfrag(brow + k0 + 16 * hi), acc);
#pragma unroll
      for (int i = 0; i < 8; ++i) {
        const int r = ti * 16 + i + 8 * hi;
        const int cidx = r * 256 + tj * 16 + ln;
        Msh[cidx] = 0.99f * Msh[cidx] + acc[i] * (1.0f / 256.0f);
      }
    }
    __syncthreads();

    // Frobenius norm clamp
    if (tid == 0) Msh[65536] = 0.0f;
    __syncthreads();
    float p = 0.0f;
    for (int i = tid; i < 65536; i += 256) { float v = Msh[i]; p += v * v; }
    atomicAdd(&Msh[65536], p);
    __syncthreads();
    const float scl = 10.0f / fmaxf(sqrtf(Msh[65536]), 10.0f);
    __syncthreads();
    for (int i = tid; i < 65536; i += 256) Msh[i] *= scl;
    __syncthreads();
  }
}

// ---------------------------------------------------------------------------
// Kernel 3: retrieval r = q_ck @ M_ck^T for all chunks in parallel.
// A tile (16x256 bf16, 8KB) staged into LDS via TDM; grid = 64 chunks * 16 tiles.
// ---------------------------------------------------------------------------
__global__ __launch_bounds__(256) void retrieve(
    const unsigned short* __restrict__ q,
    const unsigned short* __restrict__ Mstates,
    unsigned short* __restrict__ retr) {
  __shared__ unsigned short sA[16 * MD];   // 8 KB

  const int tid  = threadIdx.x;
  const int wave = tid >> 5;
  const int lane = tid & 31;
  const int hi   = lane >> 4;
  const int ln   = lane & 15;

  const int ck = blockIdx.x >> 4;
  const int rt = blockIdx.x & 15;
  const int P  = ck * 256 + rt * 16;

  stage_tile(sA, q + (size_t)P * MD, 16 * MD, tid, wave);

  const unsigned short* Ms = Mstates + (size_t)ck * 65536;
  const int n0 = wave * 32 + ln;
  const int n1 = n0 + 16;
  const unsigned short* b0 = Ms + (size_t)n0 * MD;
  const unsigned short* b1 = Ms + (size_t)n1 * MD;

  v8f acc0{}, acc1{};
  for (int k0 = 0; k0 < MD; k0 += 32) {
    v16bf a = load_afrag_bf(&sA[ln * MD + k0], hi);
    acc0 = wmma_bf16(a, load_bfrag(b0 + k0 + 16 * hi), acc0);
    acc1 = wmma_bf16(a, load_bfrag(b1 + k0 + 16 * hi), acc1);
  }
#pragma unroll
  for (int i = 0; i < 8; ++i) {
    const size_t row = (size_t)(P + i + 8 * hi) * MD;
    retr[row + n0] = f2bf_u(acc0[i]);
    retr[row + n1] = f2bf_u(acc1[i]);
  }
}

// ---------------------------------------------------------------------------
// Kernel 4: out = retrieved @ W_out^T, unmapping the token permutation.
// ---------------------------------------------------------------------------
__global__ __launch_bounds__(256) void outproj(
    const unsigned short* __restrict__ retr,
    const unsigned short* __restrict__ WoBf,
    float* __restrict__ out) {
  __shared__ unsigned short sA[16 * MD];   // 8 KB

  const int tid  = threadIdx.x;
  const int wave = tid >> 5;
  const int lane = tid & 31;
  const int hi   = lane >> 4;
  const int ln   = lane & 15;

  const int pt = blockIdx.x >> 2;
  const int cg = blockIdx.x & 3;
  const int P  = pt * 16;

  stage_tile(sA, retr + (size_t)P * MD, 16 * MD, tid, wave);

  const int h0 = cg * 256 + wave * 32 + ln;
  const int h1 = h0 + 16;
  const unsigned short* b0 = WoBf + (size_t)h0 * MD;
  const unsigned short* b1 = WoBf + (size_t)h1 * MD;

  v8f acc0{}, acc1{};
  for (int k0 = 0; k0 < MD; k0 += 32) {
    v16bf a = load_afrag_bf(&sA[ln * MD + k0], hi);
    acc0 = wmma_bf16(a, load_bfrag(b0 + k0 + 16 * hi), acc0);
    acc1 = wmma_bf16(a, load_bfrag(b1 + k0 + 16 * hi), acc1);
  }
#pragma unroll
  for (int i = 0; i < 8; ++i) {
    const int p = P + i + 8 * hi;
    const int b = (p >> 6) & 3;
    const int s = (p >> 8) * 64 + (p & 63);
    const size_t o = ((size_t)b * SS + s) * HH;
    out[o + h0] = acc0[i];
    out[o + h1] = acc1[i];
  }
}

// ---------------------------------------------------------------------------
// Host-side launch
// ---------------------------------------------------------------------------
extern "C" void kernel_launch(void* const* d_in, const int* in_sizes, int n_in,
                              void* d_out, int out_size, void* d_ws, size_t ws_size,
                              hipStream_t stream) {
  (void)in_sizes; (void)n_in; (void)out_size; (void)ws_size;
  const float* x  = (const float*)d_in[0];
  const float* Wq = (const float*)d_in[1];
  const float* Wk = (const float*)d_in[2];
  const float* Wv = (const float*)d_in[3];
  const float* Wo = (const float*)d_in[4];
  const float* M0 = (const float*)d_in[5];
  float* out = (float*)d_out;

  char* ws = (char*)d_ws;
  const size_t OFF_WQ   = 0;
  const size_t OFF_WK   = OFF_WQ + (size_t)MD * HH * 2;     //   512 KB each
  const size_t OFF_WV   = OFF_WK + (size_t)MD * HH * 2;
  const size_t OFF_WO   = OFF_WV + (size_t)MD * HH * 2;
  const size_t OFF_Q    = OFF_WO + (size_t)HH * MD * 2;
  const size_t OFF_KT   = OFF_Q    + (size_t)RTOT * MD * 2; // 8 MB each
  const size_t OFF_VT   = OFF_KT   + (size_t)MD * RTOT * 2;
  const size_t OFF_RETR = OFF_VT   + (size_t)MD * RTOT * 2;
  const size_t OFF_MST  = OFF_RETR + (size_t)RTOT * MD * 2;
  const size_t OFF_XBF  = OFF_MST  + (size_t)NCHK * MD * MD * 2; // 32 MB

  unsigned short* WqBf = (unsigned short*)(ws + OFF_WQ);
  unsigned short* WkBf = (unsigned short*)(ws + OFF_WK);
  unsigned short* WvBf = (unsigned short*)(ws + OFF_WV);
  unsigned short* WoBf = (unsigned short*)(ws + OFF_WO);
  unsigned short* qBf  = (unsigned short*)(ws + OFF_Q);
  unsigned short* kT   = (unsigned short*)(ws + OFF_KT);
  unsigned short* vT   = (unsigned short*)(ws + OFF_VT);
  unsigned short* retr = (unsigned short*)(ws + OFF_RETR);
  unsigned short* Mst  = (unsigned short*)(ws + OFF_MST);
  unsigned short* xBf  = (unsigned short*)(ws + OFF_XBF);

  const int nW = MD * HH;        // 262144
  const int nX = RTOT * HH;      // 16777216
  cvt_f32_bf16<<<nW / 1024, 256, 0, stream>>>(Wq, WqBf, nW);
  cvt_f32_bf16<<<nW / 1024, 256, 0, stream>>>(Wk, WkBf, nW);
  cvt_f32_bf16<<<nW / 1024, 256, 0, stream>>>(Wv, WvBf, nW);
  cvt_f32_bf16<<<nW / 1024, 256, 0, stream>>>(Wo, WoBf, nW);
  cvt_f32_bf16<<<nX / 1024, 256, 0, stream>>>(x, xBf, nX);

  proj_qkv<<<RTOT / 16, 256, 0, stream>>>(xBf, WqBf, WkBf, WvBf, qBf, kT, vT);

  const size_t ldsBytes = (65536 + 32) * sizeof(float);  // 256KB state + scratch
  scan_states<<<1, 256, ldsBytes, stream>>>(M0, kT, vT, Mst);

  retrieve<<<NCHK * 16, 256, 0, stream>>>(qBf, Mst, retr);

  outproj<<<(RTOT / 16) * 4, 256, 0, stream>>>(retr, WoBf, out);
}
